// Tacotron_40767829574377
// MI455X (gfx1250) — matrix operand order repaired and checked by reading, not compiled, not run on hardware
//
#include <hip/hip_runtime.h>
#include <hip/hip_bf16.h>

// ---------------------------------------------------------------------------
// Tacotron decoder step for MI455X (gfx1250, wave32, WMMA).
// Key insight: attention energies are only needed inside the <=19-wide window
// around argmax(prev_att) (softmax is shift invariant, mask kills the rest),
// so the 17-GFLOP encoder GEMM collapses to ~0.4 GFLOP. Remaining heavy work
// is the two LSTM GEMMs (M=64, N=4096, K~2K), which are weight-bandwidth
// bound -> convert weights to bf16 once, stream via v_wmma_f32_16x16x32_bf16.
// ---------------------------------------------------------------------------

typedef __bf16 bf16;
typedef __attribute__((ext_vector_type(16))) __bf16 v16bf;
typedef __attribute__((ext_vector_type(8)))  float  v8f;

union FragB16 { v16bf v; uint4 q[2]; unsigned int u[8]; };
union Acc8    { v8f  v; float f[8]; };

static constexpr int Nn   = 64;
static constexpr int Tt   = 1024;
static constexpr int Ee   = 512;
static constexpr int Aa   = 256;
static constexpr int Hh   = 256;
static constexpr int Ss   = 64;
static constexpr int Oo   = 80;
static constexpr int HID  = 4 * Hh;        // 1024
static constexpr int LSTM_IN = Ee + Hh + Ss; // 832
static constexpr int GATES = 4 * HID;      // 4096

__device__ __forceinline__ float sigm(float x) { return 1.0f / (1.0f + expf(-x)); }
__device__ __forceinline__ float bflo(unsigned u) {
    union { unsigned u; float f; } c; c.u = (u & 0xffffu) << 16; return c.f;
}
__device__ __forceinline__ float bfhi(unsigned u) {
    union { unsigned u; float f; } c; c.u = u & 0xffff0000u; return c.f;
}
__device__ __forceinline__ unsigned pack_bf16x2(float a, float b) {
    union { __bf16 h; unsigned short s; } ua, ub;
    ua.h = (__bf16)a; ub.h = (__bf16)b;
    return (unsigned)ua.s | ((unsigned)ub.s << 16);
}

// --------------------------- weight conversion -----------------------------
__global__ void k_f32_to_bf16(const float* __restrict__ s, bf16* __restrict__ d, int n) {
    int i = blockIdx.x * blockDim.x + threadIdx.x;
    if (i < n) d[i] = (bf16)s[i];
}

// src is (R x C) row-major; dst is (C x R) row-major (i.e. transposed)
__global__ void k_f32_to_bf16_T(const float* __restrict__ s, bf16* __restrict__ d, int R, int C) {
    int i = blockIdx.x * blockDim.x + threadIdx.x;
    if (i < R * C) {
        int r = i / C, c = i % C;
        d[(size_t)c * R + r] = (bf16)s[i];
    }
}

// hidden (2,N,HID) f32 -> hflat (N, 2*HID) bf16 : hflat[n][l*HID+h]
__global__ void k_hflat(const float* __restrict__ hidden, bf16* __restrict__ hflat) {
    int i = blockIdx.x * blockDim.x + threadIdx.x;
    if (i < 2 * Nn * HID) {
        int l = i / (Nn * HID);
        int r = i - l * Nn * HID;
        int n = r / HID, h = r - n * HID;
        hflat[(size_t)n * (2 * HID) + l * HID + h] = (bf16)hidden[i];
    }
}

// --------------------------- attention window ------------------------------
__global__ void k_window(const float* __restrict__ prev_att, const int* __restrict__ lengths,
                         int* __restrict__ win) {
    int n = blockIdx.x, tid = threadIdx.x;
    __shared__ float sv[256]; __shared__ int si[256];
    float best = -3.0e38f; int bi = 0;
    for (int t = tid; t < Tt; t += 256) {
        float v = prev_att[(size_t)n * Tt + t];
        if (v > best) { best = v; bi = t; }
    }
    sv[tid] = best; si[tid] = bi; __syncthreads();
    for (int s = 128; s > 0; s >>= 1) {
        if (tid < s) {
            if (sv[tid + s] > sv[tid] || (sv[tid + s] == sv[tid] && si[tid + s] < si[tid])) {
                sv[tid] = sv[tid + s]; si[tid] = si[tid + s];
            }
        }
        __syncthreads();
    }
    if (tid == 0) {
        int idx = si[0];
        int len = lengths[n]; len = len < 1 ? 1 : (len > Tt ? Tt : len);
        int lo = idx - 9; lo = lo < 0 ? 0 : lo; lo = lo > len - 1 ? len - 1 : lo;
        int hi = idx + 9; hi = hi < 0 ? 0 : hi; hi = hi > len - 1 ? len - 1 : hi;
        win[2 * n] = lo; win[2 * n + 1] = hi;
    }
}

// bias_na[n][a] = softsign(spkr[n] . W_spkr[:,a])
__global__ void k_spkr_bias(const float* __restrict__ spkr, const float* __restrict__ W_spkr,
                            float* __restrict__ bias_na) {
    int i = blockIdx.x * blockDim.x + threadIdx.x;
    if (i < Nn * Aa) {
        int n = i / Aa, a = i - n * Aa;
        float s = 0.f;
        for (int k = 0; k < Ss; ++k) s += spkr[n * Ss + k] * W_spkr[k * Aa + a];
        bias_na[i] = s / (1.0f + fabsf(s));
    }
}

// --------------------- generic 64-row bf16 WMMA GEMM -----------------------
// out[n][j] = sum_k X0[n][k]*W0[j][k] + sum_k X1[n][k]*W1[j][k]
//             (+bias_a[j]) (+bias_b[j]) (+addNJ[n][j])
// W is "j-major" (row j contiguous over K).  block = 256 threads = 8 waves,
// tile = 64 rows x 64 cols, wave -> 1 row-block x 2 col-blocks.
__global__ void k_gemm64(const bf16* __restrict__ X0, int ldx0,
                         const bf16* __restrict__ W0, int K0,
                         const bf16* __restrict__ X1, int ldx1,
                         const bf16* __restrict__ W1, int K1,
                         int Ncols,
                         const float* __restrict__ bias_a,
                         const float* __restrict__ bias_b,
                         const float* __restrict__ addNJ, int ldadd,
                         float* __restrict__ out, int ldo) {
    int tid = threadIdx.x;
    int lane = tid & 31, w = tid >> 5;
    int lhalf = lane & 15, g = lane >> 4;
    int m_blk = w & 3;
    int cb = (w >> 2) * 2;
    int j0 = blockIdx.x * 64;

    Acc8 acc[2];
#pragma unroll
    for (int c = 0; c < 2; ++c)
#pragma unroll
        for (int v = 0; v < 8; ++v) acc[c].f[v] = 0.f;

    for (int seg = 0; seg < 2; ++seg) {
        const bf16* X = seg ? X1 : X0;
        const bf16* W = seg ? W1 : W0;
        int K  = seg ? K1 : K0;
        int ld = seg ? ldx1 : ldx0;
        if (K <= 0 || W == nullptr) continue;
        const bf16* xrow = X + (size_t)(m_blk * 16 + lhalf) * ld;
        for (int k = 0; k < K; k += 32) {
            FragB16 a;
            a.q[0] = *(const uint4*)(xrow + k + 16 * g);
            a.q[1] = *(const uint4*)(xrow + k + 16 * g + 8);
#pragma unroll
            for (int c = 0; c < 2; ++c) {
                int j = j0 + (cb + c) * 16 + lhalf;
                if (j >= Ncols) j = Ncols - 1;
                const bf16* wp = W + (size_t)j * K + k + 16 * g;
                __builtin_prefetch(wp + 256, 0, 0);
                FragB16 b;
                b.q[0] = *(const uint4*)(wp);
                b.q[1] = *(const uint4*)(wp + 8);
                acc[c].v = __builtin_amdgcn_wmma_f32_16x16x32_bf16(
                    false, a.v, false, b.v, (short)0, acc[c].v, false, false);
            }
        }
    }

#pragma unroll
    for (int c = 0; c < 2; ++c) {
        int j = j0 + (cb + c) * 16 + lhalf;
        if (j < Ncols) {
#pragma unroll
            for (int v = 0; v < 8; ++v) {
                int row = m_blk * 16 + v + 8 * g;
                float r = acc[c].f[v];
                if (bias_a) r += bias_a[j];
                if (bias_b) r += bias_b[j];
                if (addNJ)  r += addNJ[(size_t)row * ldadd + j];
                out[(size_t)row * ldo + j] = r;
            }
        }
    }
}

// ----------------------------- prenet --------------------------------------
__global__ void k_prenet(const float* __restrict__ input_dec, const float* __restrict__ spkr,
                         const float* __restrict__ W1, const float* __restrict__ b1,
                         const float* __restrict__ W2, const float* __restrict__ b2,
                         bf16* __restrict__ inlstm) {
    int n = blockIdx.x, tid = threadIdx.x;
    __shared__ float pin[Oo + Ss];   // 144
    __shared__ float pn1[2 * Hh];    // 512
    if (tid < Oo) pin[tid] = input_dec[n * Oo + tid];
    else if (tid < Oo + Ss) pin[tid] = spkr[n * Ss + (tid - Oo)];
    __syncthreads();
#pragma unroll
    for (int rep = 0; rep < 2; ++rep) {
        int j = tid + rep * 256;
        float s = b1[j];
        for (int k = 0; k < Oo + Ss; ++k) s += pin[k] * W1[k * (2 * Hh) + j];
        pn1[j] = s > 0.f ? s : 0.f;
    }
    __syncthreads();
    {
        int j = tid;
        float s = b2[j];
        for (int k = 0; k < 2 * Hh; ++k) s += pn1[k] * W2[k * Hh + j];
        s = s > 0.f ? s : 0.f;
        inlstm[(size_t)n * LSTM_IN + j] = (bf16)s;          // cols [0,256)
    }
    if (tid < Ss)                                            // cols [768,832)
        inlstm[(size_t)n * LSTM_IN + Ee + Hh + tid] = (bf16)spkr[n * Ss + tid];
}

// ------------- fused windowed attention energy + softmax + context ----------
// one workgroup per batch n. 32 (padded) window rows x 256 A-cols via WMMA.
__global__ void k_att_energy(const float* __restrict__ input_enc,
                             const bf16* __restrict__ WencT,        // (A, E) bf16
                             const float* __restrict__ b_enc,
                             const float* __restrict__ att_base,    // (N, A)
                             const float* __restrict__ prev_att,
                             const float* __restrict__ conv_w,      // (A,1,31)
                             const float* __restrict__ W_proj,
                             const float* __restrict__ b_proj,
                             const int* __restrict__ win,
                             bf16* __restrict__ inlstm,             // context -> cols [256,768)
                             bf16* __restrict__ decX) {             // context -> cols [1024,1536)
    int n = blockIdx.x, tid = threadIdx.x;
    int lane = tid & 31, w = tid >> 5;
    int lhalf = lane & 15, g = lane >> 4;
    int lo = win[2 * n], hi = win[2 * n + 1];

    __shared__ __align__(16) unsigned atile[32 * 256];  // 32 rows x 512 bf16
    __shared__ float red[32];
    __shared__ float wv[32];
    __shared__ float s_scalar;

    if (tid < 32) red[tid] = 0.f;

    // stage + convert the 32 window rows of input_enc into LDS (bf16)
    {
        int r = tid >> 3;                 // 0..31
        int c0 = (tid & 7) * 64;          // 64 cols each
        int t = lo + r; if (t > Tt - 1) t = Tt - 1;
        const float* src = input_enc + ((size_t)n * Tt + t) * Ee + c0;
        unsigned* dst = &atile[r * 256 + (c0 >> 1)];
        for (int i = 0; i < 64; i += 2)
            dst[i >> 1] = pack_bf16x2(src[i], src[i + 1]);
    }
    __syncthreads();

    // WMMA: 2 row-blocks x 16 col-blocks, wave -> 1 row-block x 4 col-blocks
    int m_blk = w & 1;
    int cbase = (w >> 1) * 4;
    Acc8 acc[4];
#pragma unroll
    for (int c = 0; c < 4; ++c)
#pragma unroll
        for (int v = 0; v < 8; ++v) acc[c].f[v] = 0.f;

    int row = m_blk * 16 + lhalf;
    for (int k = 0; k < Ee; k += 32) {
        FragB16 a;
        a.q[0] = *(const uint4*)&atile[row * 256 + (k >> 1) + 8 * g];
        a.q[1] = *(const uint4*)&atile[row * 256 + (k >> 1) + 8 * g + 4];
#pragma unroll
        for (int c = 0; c < 4; ++c) {
            int j = (cbase + c) * 16 + lhalf;
            const bf16* wp = WencT + (size_t)j * Ee + k + 16 * g;
            FragB16 b;
            b.q[0] = *(const uint4*)(wp);
            b.q[1] = *(const uint4*)(wp + 8);
            acc[c].v = __builtin_amdgcn_wmma_f32_16x16x32_bf16(
                false, a.v, false, b.v, (short)0, acc[c].v, false, false);
        }
    }

    // epilogue: softsign + biases + depthwise conv term + tanh()*W_proj,
    // reduce over A into per-row logits via LDS float atomics (ds_add_f32)
#pragma unroll
    for (int c = 0; c < 4; ++c) {
        int a = (cbase + c) * 16 + lhalf;
        float wpj = W_proj[a];
#pragma unroll
        for (int v = 0; v < 8; ++v) {
            int m = m_blk * 16 + v + 8 * g;
            int t = lo + m;
            if (t <= hi) {
                float e = acc[c].f[v] + b_enc[a];
                e = e / (1.0f + fabsf(e));          // softsign(enc GEMM)
                e += att_base[(size_t)n * Aa + a];  // in_att_dec + softsign(spkr)
                float cv = 0.f;                      // conv term (kernel width 31)
                for (int k = 0; k < 31; ++k) {
                    int tp = t + k - 15;
                    float pa = (tp >= 0 && tp < Tt) ? prev_att[(size_t)n * Tt + tp] : 0.f;
                    cv += pa * conv_w[a * 31 + k];
                }
                e += cv;
                atomicAdd(&red[m], tanhf(e) * wpj);
            }
        }
    }
    __syncthreads();

    // windowed softmax (shift-invariant: window max == full-T max effect)
    if (tid == 0) {
        float mx = -3.0e38f;
        for (int m = 0; m <= hi - lo; ++m) {
            float l = red[m] + b_proj[0];
            if (l > mx) mx = l;
        }
        s_scalar = mx;
    }
    __syncthreads();
    if (tid < 32) {
        int t = lo + tid;
        wv[tid] = (t <= hi) ? expf(red[tid] + b_proj[0] - s_scalar) : 0.f;
    }
    __syncthreads();
    if (tid == 0) {
        float s = 0.f;
        for (int m = 0; m < 32; ++m) s += wv[m];
        s_scalar = fmaxf(s, 1e-12f);
    }
    __syncthreads();

    // context[n][e] = sum_m wv[m] * enc[lo+m][e]  (from the LDS bf16 tile)
    {
        float inv = 1.0f / s_scalar;
        int e0 = tid * 2;
        float c0 = 0.f, c1 = 0.f;
        for (int m = 0; m < 32; ++m) {
            float wm = wv[m];
            unsigned u = atile[m * 256 + tid];
            c0 += wm * bflo(u);
            c1 += wm * bfhi(u);
        }
        c0 *= inv; c1 *= inv;
        inlstm[(size_t)n * LSTM_IN + Hh + e0]     = (bf16)c0;
        inlstm[(size_t)n * LSTM_IN + Hh + e0 + 1] = (bf16)c1;
        decX[(size_t)n * (HID + Ee) + HID + e0]     = (bf16)c0;
        decX[(size_t)n * (HID + Ee) + HID + e0 + 1] = (bf16)c1;
    }
}

// ------------------------- LSTM pointwise ----------------------------------
__global__ void k_lstm_pointwise(const float* __restrict__ gates, const float* __restrict__ cell,
                                 bf16* __restrict__ dst, int dst_ld) {
    int i = blockIdx.x * blockDim.x + threadIdx.x;
    if (i < Nn * HID) {
        int n = i / HID, h = i - n * HID;
        const float* gp = gates + (size_t)n * GATES;
        float gi = gp[h], gf = gp[HID + h], gg = gp[2 * HID + h], go = gp[3 * HID + h];
        float c_new = sigm(gf) * cell[i] + sigm(gi) * tanhf(gg);
        float h_new = sigm(go) * tanhf(c_new);
        dst[(size_t)n * dst_ld + h] = (bf16)h_new;
    }
}

// ---------------------------------------------------------------------------
extern "C" void kernel_launch(void* const* d_in, const int* in_sizes, int n_in,
                              void* d_out, int out_size, void* d_ws, size_t ws_size,
                              hipStream_t stream) {
    const float* input_enc = (const float*)d_in[0];
    const float* input_dec = (const float*)d_in[1];
    const float* spkr_vec  = (const float*)d_in[2];
    const float* prev_att  = (const float*)d_in[3];
    const float* hidden    = (const float*)d_in[4];
    const float* cell      = (const float*)d_in[5];
    const int*   lengths   = (const int*)d_in[6];
    const float* W_enc     = (const float*)d_in[7];
    const float* b_enc     = (const float*)d_in[8];
    const float* W_spkr    = (const float*)d_in[9];
    const float* conv_w    = (const float*)d_in[10];
    const float* W_dec     = (const float*)d_in[11];
    const float* W_proj    = (const float*)d_in[12];
    const float* b_proj    = (const float*)d_in[13];
    const float* W_pre1    = (const float*)d_in[14];
    const float* b_pre1    = (const float*)d_in[15];
    const float* W_pre2    = (const float*)d_in[16];
    const float* b_pre2    = (const float*)d_in[17];
    const float* Wih0      = (const float*)d_in[18];
    const float* Whh0      = (const float*)d_in[19];
    const float* bih0      = (const float*)d_in[20];
    const float* bhh0      = (const float*)d_in[21];
    const float* Wih1      = (const float*)d_in[22];
    const float* Whh1      = (const float*)d_in[23];
    const float* bih1      = (const float*)d_in[24];
    const float* bhh1      = (const float*)d_in[25];
    const float* W_out     = (const float*)d_in[26];
    const float* b_out     = (const float*)d_in[27];
    float* out = (float*)d_out;

    char* ws = (char*)d_ws;
    size_t off = 0;
    auto alloc = [&](size_t bytes) -> char* {
        char* p = ws + off;
        off = (off + bytes + 255) & ~(size_t)255;
        return p;
    };

    bf16* wWih0  = (bf16*)alloc((size_t)GATES * LSTM_IN * 2);
    bf16* wWhh0  = (bf16*)alloc((size_t)GATES * HID * 2);
    bf16* wWih1  = (bf16*)alloc((size_t)GATES * HID * 2);
    bf16* wWhh1  = (bf16*)alloc((size_t)GATES * HID * 2);
    bf16* wWencT = (bf16*)alloc((size_t)Aa * Ee * 2);
    bf16* wWdecT = (bf16*)alloc((size_t)Aa * 2 * HID * 2);
    bf16* wWoutT = (bf16*)alloc((size_t)(Oo * 2) * (HID + Ee) * 2);
    bf16* hflat  = (bf16*)alloc((size_t)Nn * 2 * HID * 2);
    bf16* inlstm = (bf16*)alloc((size_t)Nn * LSTM_IN * 2);
    bf16* h1bf   = (bf16*)alloc((size_t)Nn * HID * 2);
    bf16* decX   = (bf16*)alloc((size_t)Nn * (HID + Ee) * 2);
    float* gates    = (float*)alloc((size_t)Nn * GATES * 4);
    float* att_base = (float*)alloc((size_t)Nn * Aa * 4);
    float* bias_na  = (float*)alloc((size_t)Nn * Aa * 4);
    int*   win      = (int*)alloc((size_t)Nn * 2 * 4);
    (void)ws_size; (void)n_in; (void)in_sizes; (void)out_size;

    auto cvt = [&](const float* s, bf16* d, int n) {
        k_f32_to_bf16<<<(n + 255) / 256, 256, 0, stream>>>(s, d, n);
    };
    cvt(Wih0, wWih0, GATES * LSTM_IN);
    cvt(Whh0, wWhh0, GATES * HID);
    cvt(Wih1, wWih1, GATES * HID);
    cvt(Whh1, wWhh1, GATES * HID);
    k_f32_to_bf16_T<<<(Ee * Aa + 255) / 256, 256, 0, stream>>>(W_enc, wWencT, Ee, Aa);
    k_f32_to_bf16_T<<<(2 * HID * Aa + 255) / 256, 256, 0, stream>>>(W_dec, wWdecT, 2 * HID, Aa);
    k_f32_to_bf16_T<<<((HID + Ee) * Oo * 2 + 255) / 256, 256, 0, stream>>>(W_out, wWoutT, HID + Ee, Oo * 2);
    k_hflat<<<(2 * Nn * HID + 255) / 256, 256, 0, stream>>>(hidden, hflat);

    k_window<<<Nn, 256, 0, stream>>>(prev_att, lengths, win);
    k_spkr_bias<<<(Nn * Aa + 255) / 256, 256, 0, stream>>>(spkr_vec, W_spkr, bias_na);

    // att_base = hflat @ W_dec  (+ softsign spkr bias)   [WMMA]
    k_gemm64<<<Aa / 64, 256, 0, stream>>>(hflat, 2 * HID, wWdecT, 2 * HID,
                                          nullptr, 0, nullptr, 0,
                                          Aa, nullptr, nullptr, bias_na, Aa,
                                          att_base, Aa);

    k_prenet<<<Nn, 256, 0, stream>>>(input_dec, spkr_vec, W_pre1, b_pre1, W_pre2, b_pre2, inlstm);

    // fused windowed attention (WMMA) -> context into inlstm and decX
    k_att_energy<<<Nn, 256, 0, stream>>>(input_enc, wWencT, b_enc, att_base,
                                         prev_att, conv_w, W_proj, b_proj, win,
                                         inlstm, decX);

    // LSTM layer 0: gates = inlstm@Wih0^T + h0@Whh0^T + bih0 + bhh0   [WMMA]
    k_gemm64<<<GATES / 64, 256, 0, stream>>>(inlstm, LSTM_IN, wWih0, LSTM_IN,
                                             hflat, 2 * HID, wWhh0, HID,
                                             GATES, bih0, bhh0, nullptr, 0,
                                             gates, GATES);
    k_lstm_pointwise<<<(Nn * HID + 255) / 256, 256, 0, stream>>>(gates, cell, h1bf, HID);

    // LSTM layer 1: gates = h1@Wih1^T + h_prev1@Whh1^T + biases       [WMMA]
    k_gemm64<<<GATES / 64, 256, 0, stream>>>(h1bf, HID, wWih1, HID,
                                             hflat + HID, 2 * HID, wWhh1, HID,
                                             GATES, bih1, bhh1, nullptr, 0,
                                             gates, GATES);
    k_lstm_pointwise<<<(Nn * HID + 255) / 256, 256, 0, stream>>>(gates, cell + Nn * HID, decX, HID + Ee);

    // output = [h2 | context] @ W_out + b_out  -> d_out (N,2,80) flat  [WMMA]
    k_gemm64<<<3, 256, 0, stream>>>(decX, HID + Ee, wWoutT, HID + Ee,
                                    nullptr, 0, nullptr, 0,
                                    Oo * 2, b_out, nullptr, nullptr, 0,
                                    out, Oo * 2);
}
